// TimeThenSpaceModel_26207890440479
// MI455X (gfx1250) — compile-verified, matrix operand order checked by reading, not run on hardware
//
#include <hip/hip_runtime.h>

typedef __attribute__((ext_vector_type(16))) _Float16 v16h;
typedef __attribute__((ext_vector_type(8)))  float    v8f;
typedef _Float16 half_t;

#define Bc    8
#define Tc    12
#define Nc    10000
#define Fc    8
#define Hc    64
#define Gc    192      // 3*H
#define Ec    320000
#define Kcat  320      // (2K+1)*H
#define Jc    96       // F*HOR
#define Mc    (Bc*Nc)  // 80000
#define TILES (Mc/16)  // 5000  (Nc%16==0 -> no tile straddles a batch boundary)
#define GRU_WAVES 4

// fast activations: single v_rcp_f32 instead of IEEE divide expansion
__device__ __forceinline__ float fast_sigmoid(float v) {
  return __builtin_amdgcn_rcpf(1.f + __expf(-v));
}
__device__ __forceinline__ float fast_tanh(float v) {
  return 2.f * __builtin_amdgcn_rcpf(1.f + __expf(-2.f * v)) - 1.f;
}

// ---- WMMA fragment loader (ISA 7.12.2 layout, wave32) ----
// A-matrix 16x32 f16: lane L (L<16) row M=L holds K {0..7,16..23}; lane>=16 row M=L-16 holds
// K {8..15,24..31}. Also loads B fragments when B is stored n-major: B[k][n] at base[n*stride+k].
// Each fragment = two runs of 8 contiguous halves per lane -> 2x global_load_b128.
__device__ __forceinline__ v16h load_frag_nmajor_h(const half_t* __restrict__ base,
                                                   int row_stride, int row0, int k0) {
  int lane = threadIdx.x & 31;
  const half_t* p = base + (size_t)(row0 + (lane & 15)) * row_stride + k0;
  int khalf = (lane >> 4) * 8;
  v16h a;
#pragma unroll
  for (int i = 0; i < 16; ++i) {
    int k = khalf + (i < 8 ? i : i + 8);
    a[i] = p[k];
  }
  return a;
}

// ---- weight prep: Whh->f16 (g-major), W1h = (enc_W @ Wih^T)^T zero-padded to [192][32] f16,
//      Wfd^T = (filt_W @ dec_W)^T as [96][320] f16, bias_fd ----
__global__ void prep_weights_k(const float* __restrict__ Whh, const float* __restrict__ enc_W,
                               const float* __restrict__ Wih, const float* __restrict__ filt_W,
                               const float* __restrict__ dec_W, const float* __restrict__ filt_b,
                               const float* __restrict__ dec_b,
                               half_t* __restrict__ Whh_h, half_t* __restrict__ W1h,
                               half_t* __restrict__ Wfd_h, float* __restrict__ bias_fd) {
  int tid = blockIdx.x * blockDim.x + threadIdx.x;
  if (tid < Gc * Hc) { Whh_h[tid] = (half_t)Whh[tid]; return; }
  tid -= Gc * Hc;
  if (tid < Gc * 32) {          // W1h[g][k] = sum_h enc_W[k,h] * Wih[g,h] for k<8, else 0
    int g = tid / 32, k = tid % 32;
    half_t v = (half_t)0.f;
    if (k < Fc) {
      float s = 0.f;
#pragma unroll
      for (int h = 0; h < Hc; ++h) s += enc_W[k * Hc + h] * Wih[g * Hc + h];
      v = (half_t)s;
    }
    W1h[tid] = v;
    return;
  }
  tid -= Gc * 32;
  if (tid < Jc * Kcat) {        // Wfd_h[j][k] = sum_h filt_W[k,h] * dec_W[h,j]  (n-major)
    int j = tid / Kcat, k = tid % Kcat;
    float s = 0.f;
#pragma unroll
    for (int h = 0; h < Hc; ++h) s += filt_W[k * Hc + h] * dec_W[h * Jc + j];
    Wfd_h[tid] = (half_t)s;
    return;
  }
  tid -= Jc * Kcat;
  if (tid < Jc) {               // bias_fd[j] = filt_b @ dec_W + dec_b
    float s = dec_b[tid];
#pragma unroll
    for (int h = 0; h < Hc; ++h) s += filt_b[h] * dec_W[h * Jc + tid];
    bias_fd[tid] = s;
  }
}

// bias_node[n,g] = bih[g] + sum_h (node_emb[n,h]+enc_b[h]) * Wih[g,h]  (+ bhh[g] for r/z gates)
__global__ void bias_node_k(const float* __restrict__ node_emb, const float* __restrict__ enc_b,
                            const float* __restrict__ Wih, const float* __restrict__ bih,
                            const float* __restrict__ bhh, float* __restrict__ bias_node) {
  int tid = blockIdx.x * blockDim.x + threadIdx.x;
  if (tid >= Nc * Gc) return;
  int n = tid / Gc, g = tid % Gc;
  float s = bih[g];
  if (g < 128) s += bhh[g];     // bhh folds additively for r,z gates only (n gate scales by r)
#pragma unroll
  for (int h = 0; h < Hc; ++h) s += (node_emb[n * Hc + h] + enc_b[h]) * Wih[g * Hc + h];
  bias_node[tid] = s;
}

// ---- GRU, all 12 timesteps in one kernel. The recurrence is row-local, so a 16-row tile
//      never leaves the wave: h lives in f32 registers (C layout) + an f16 LDS shadow that is
//      re-read in A-fragment layout each step. Wave-private LDS slices -> no barriers. ----
__global__ void __launch_bounds__(GRU_WAVES * 32)
gru_all_k(const float* __restrict__ x, const half_t* __restrict__ W1h,
          const float* __restrict__ bias_node, const float* __restrict__ bhh,
          const half_t* __restrict__ Whh_h, float* __restrict__ h_out) {
  __shared__ half_t hsh[GRU_WAVES][16][72];    // padded rows (bank spread)
  __shared__ float  bns[GRU_WAVES][16][196];   // bias_node tile rows, padded

  int wave = threadIdx.x >> 5;
  int lane = threadIdx.x & 31;
  int tile = blockIdx.x * GRU_WAVES + wave;
  int m0 = tile * 16;
  int bb = m0 / Nc;             // wave-uniform: tile never straddles a batch (Nc%16==0)
  int n0 = m0 - bb * Nc;

  // zero h shadow (h0 = 0)
  {
    unsigned* hw = (unsigned*)&hsh[wave][0][0];   // 16*72/2 = 576 dwords
    for (int i = lane; i < 576; i += 32) hw[i] = 0u;
  }
  // stage this tile's bias rows into LDS (float4 copies; 196 cols = 49 float4s)
  {
    const float4* src = (const float4*)(bias_node + (size_t)n0 * Gc);
    float4* dst = (float4*)&bns[wave][0][0];
    for (int i = lane; i < 16 * 48; i += 32) {
      int row = i / 48, c4 = i % 48;
      dst[row * 49 + c4] = src[row * 48 + c4];
    }
  }

  int rbase = (lane >> 4) * 8;
  int col = lane & 15;
  int rsel = lane & 15;
  int kh = (lane >> 4) * 8;

  float hreg[4][8];             // h in C layout: [j][r] -> (row rbase+r, col col+16j)
#pragma unroll
  for (int j = 0; j < 4; ++j)
#pragma unroll
    for (int r = 0; r < 8; ++r) hreg[j][r] = 0.f;

  float bhn_reg[4];
#pragma unroll
  for (int j = 0; j < 4; ++j) bhn_reg[j] = bhh[128 + col + 16 * j];

  const float* xrow0 = x + (((size_t)bb * Tc) * Nc + (n0 + lane)) * Fc;  // valid for lane<16
  v16h ax = {};                 // K=0..7 live entirely in lanes 0-15; rest stays zero

  for (int t = 0; t < Tc; ++t) {
    // A fragments from the f16 shadow (same-wave DS ordering makes step t-1 stores visible)
    v16h a0, a1;
#pragma unroll
    for (int i = 0; i < 16; ++i) {
      int k = kh + (i < 8 ? i : i + 8);
      a0[i] = hsh[wave][rsel][k];
      a1[i] = hsh[wave][rsel][32 + k];
    }
    if (lane < 16) {
      const float* xr = xrow0 + (size_t)t * Nc * Fc;
#pragma unroll
      for (int f = 0; f < Fc; ++f) ax[f] = (half_t)xr[f];
    }

#pragma unroll
    for (int j = 0; j < 4; ++j) {
      // r gate (tile j), z gate (tile 4+j): gx and gh share one accumulator
      v8f cr = {}, cz = {}, chn = {}, cxn = {};
      {
        v16h b0 = load_frag_nmajor_h(Whh_h, Hc, j * 16, 0);
        v16h b1 = load_frag_nmajor_h(Whh_h, Hc, j * 16, 32);
        v16h bx = load_frag_nmajor_h(W1h, 32, j * 16, 0);
        cr = __builtin_amdgcn_wmma_f32_16x16x32_f16(false, a0, false, b0, (short)0, cr, false, false);
        cr = __builtin_amdgcn_wmma_f32_16x16x32_f16(false, a1, false, b1, (short)0, cr, false, false);
        cr = __builtin_amdgcn_wmma_f32_16x16x32_f16(false, ax, false, bx, (short)0, cr, false, false);
      }
      {
        v16h b0 = load_frag_nmajor_h(Whh_h, Hc, (4 + j) * 16, 0);
        v16h b1 = load_frag_nmajor_h(Whh_h, Hc, (4 + j) * 16, 32);
        v16h bx = load_frag_nmajor_h(W1h, 32, (4 + j) * 16, 0);
        cz = __builtin_amdgcn_wmma_f32_16x16x32_f16(false, a0, false, b0, (short)0, cz, false, false);
        cz = __builtin_amdgcn_wmma_f32_16x16x32_f16(false, a1, false, b1, (short)0, cz, false, false);
        cz = __builtin_amdgcn_wmma_f32_16x16x32_f16(false, ax, false, bx, (short)0, cz, false, false);
      }
      // n gate (tile 8+j): gh and gx must stay separate (tanh(xn + r*hn))
      {
        v16h b0 = load_frag_nmajor_h(Whh_h, Hc, (8 + j) * 16, 0);
        v16h b1 = load_frag_nmajor_h(Whh_h, Hc, (8 + j) * 16, 32);
        v16h bx = load_frag_nmajor_h(W1h, 32, (8 + j) * 16, 0);
        chn = __builtin_amdgcn_wmma_f32_16x16x32_f16(false, a0, false, b0, (short)0, chn, false, false);
        chn = __builtin_amdgcn_wmma_f32_16x16x32_f16(false, a1, false, b1, (short)0, chn, false, false);
        cxn = __builtin_amdgcn_wmma_f32_16x16x32_f16(false, ax, false, bx, (short)0, cxn, false, false);
      }

      int hcol = col + 16 * j;
#pragma unroll
      for (int r = 0; r < 8; ++r) {
        int rl = rbase + r;
        float rg = fast_sigmoid(cr[r] + bns[wave][rl][hcol]);
        float zg = fast_sigmoid(cz[r] + bns[wave][rl][64 + hcol]);
        float xn = cxn[r] + bns[wave][rl][128 + hcol];
        float hn = chn[r] + bhn_reg[j];
        float ng = fast_tanh(xn + rg * hn);
        float hv = (1.f - zg) * ng + zg * hreg[j][r];
        hreg[j][r] = hv;
        hsh[wave][rl][hcol] = (half_t)hv;
      }
    }
  }

  // final hidden state to global (f32) for the diffusion-conv stage
#pragma unroll
  for (int j = 0; j < 4; ++j) {
    int hcol = col + 16 * j;
#pragma unroll
    for (int r = 0; r < 8; ++r)
      h_out[(size_t)(m0 + rbase + r) * Hc + hcol] = hreg[j][r];
  }
}

// ---- diffusion conv: degree, normalized weights, scatter propagation ----
__global__ void deg_k(const int* __restrict__ src, const int* __restrict__ tgt,
                      const float* __restrict__ w, float* __restrict__ deg_f,
                      float* __restrict__ deg_b) {
  int e = blockIdx.x * blockDim.x + threadIdx.x;
  if (e >= Ec) return;
  atomicAdd(&deg_f[tgt[e]], w[e]);
  atomicAdd(&deg_b[src[e]], w[e]);
}

__global__ void nw_k(const int* __restrict__ src, const int* __restrict__ tgt,
                     const float* __restrict__ w, const float* __restrict__ deg_f,
                     const float* __restrict__ deg_b, float* __restrict__ nw_f,
                     float* __restrict__ nw_b) {
  int e = blockIdx.x * blockDim.x + threadIdx.x;
  if (e >= Ec) return;
  float df = deg_f[tgt[e]], db = deg_b[src[e]];
  nw_f[e] = w[e] / (df == 0.f ? 1.f : df);
  nw_b[e] = w[e] / (db == 0.f ? 1.f : db);
}

// out[b, t_e, :] += in[b, s_e, :] * nw[e]; thread = (edge, batch, 16-wide h chunk)
__global__ void prop_k(const int* __restrict__ sidx, const int* __restrict__ tidx,
                       const float* __restrict__ nw, const float* __restrict__ in,
                       float* __restrict__ out) {
  int tid = blockIdx.x * blockDim.x + threadIdx.x;
  if (tid >= Ec * 32) return;
  int e = tid >> 5;
  int sub = tid & 31;
  int b = sub >> 2;
  int q = sub & 3;
  int s = sidx[e], t = tidx[e];
  float w = nw[e];
  const float4* ip = (const float4*)(in + ((size_t)b * Nc + s) * Hc + q * 16);
  float* op = out + ((size_t)b * Nc + t) * Hc + q * 16;
#pragma unroll
  for (int v = 0; v < 4; ++v) {
    float4 d = ip[v];
    atomicAdd(op + v * 4 + 0, d.x * w);
    atomicAdd(op + v * 4 + 1, d.y * w);
    atomicAdd(op + v * 4 + 2, d.z * w);
    atomicAdd(op + v * 4 + 3, d.w * w);
  }
}

// ---- fused tail GEMM: [h|z1|z2|z3|z4] @ Wfd + bias_fd, scattered to [B,HOR,N,F] ----
__global__ void __launch_bounds__(GRU_WAVES * 32)
out_gemm_k(const float* __restrict__ p0, const float* __restrict__ p1,
           const float* __restrict__ p2, const float* __restrict__ p3,
           const float* __restrict__ p4, const half_t* __restrict__ Wfd_h,
           const float* __restrict__ bias_fd, float* __restrict__ out) {
  int wave = threadIdx.x >> 5;
  int lane = threadIdx.x & 31;
  int tile = blockIdx.x * GRU_WAVES + wave;
  int m0 = tile * 16;
  int bb = m0 / Nc;             // wave-uniform
  int n0 = m0 - bb * Nc;
  const float* parts[5] = {p0, p1, p2, p3, p4};

  int row = m0 + (lane & 15);
  int khalf = (lane >> 4) * 8;

  v16h a[10];
#pragma unroll
  for (int kt = 0; kt < 10; ++kt) {
    int part = kt >> 1;
    int k0 = (kt & 1) * 32;
    const float* base = parts[part] + (size_t)row * Hc + k0;
#pragma unroll
    for (int i = 0; i < 16; ++i) {
      int k = khalf + (i < 8 ? i : i + 8);
      a[kt][i] = (half_t)base[k];
    }
  }

#pragma unroll
  for (int nt = 0; nt < Jc / 16; ++nt) {
    v8f c = {};
#pragma unroll
    for (int kt = 0; kt < 10; ++kt) {
      v16h bfrag = load_frag_nmajor_h(Wfd_h, Kcat, nt * 16, kt * 32);
      c = __builtin_amdgcn_wmma_f32_16x16x32_f16(false, a[kt], false, bfrag, (short)0, c,
                                                 false, false);
    }
    int jcol = nt * 16 + (lane & 15);
    int thor = jcol / Fc;
    int f = jcol % Fc;
    float bias = bias_fd[jcol];
    int rbase = (lane >> 4) * 8;
#pragma unroll
    for (int r = 0; r < 8; ++r) {
      int n = n0 + rbase + r;
      out[(((size_t)bb * Tc + thor) * Nc + n) * Fc + f] = c[r] + bias;
    }
  }
}

extern "C" void kernel_launch(void* const* d_in, const int* in_sizes, int n_in,
                              void* d_out, int out_size, void* d_ws, size_t ws_size,
                              hipStream_t stream) {
  const float* x         = (const float*)d_in[0];
  const int*   edge_idx  = (const int*)d_in[1];
  const float* edge_w    = (const float*)d_in[2];
  const float* enc_W     = (const float*)d_in[3];
  const float* enc_b     = (const float*)d_in[4];
  const float* node_emb  = (const float*)d_in[5];
  const float* Wih       = (const float*)d_in[6];
  const float* Whh       = (const float*)d_in[7];
  const float* bih       = (const float*)d_in[8];
  const float* bhh       = (const float*)d_in[9];
  const float* filt_W    = (const float*)d_in[10];
  const float* filt_b    = (const float*)d_in[11];
  const float* dec_W     = (const float*)d_in[12];
  const float* dec_b     = (const float*)d_in[13];
  const int* e_src = edge_idx;
  const int* e_tgt = edge_idx + Ec;

  char* w = (char*)d_ws;
  size_t off = 0;
  auto alloc = [&](size_t bytes) -> size_t {
    size_t o = off;
    off += (bytes + 255) & ~(size_t)255;
    return o;
  };
  size_t o_h    = alloc((size_t)Mc * Hc * 4);
  size_t o_z1   = alloc((size_t)Mc * Hc * 4);
  size_t o_z2   = alloc((size_t)Mc * Hc * 4);
  size_t o_z3   = alloc((size_t)Mc * Hc * 4);
  size_t o_z4   = alloc((size_t)Mc * Hc * 4);
  size_t o_WhhH = alloc((size_t)Gc * Hc * 2);
  size_t o_W1h  = alloc((size_t)Gc * 32 * 2);
  size_t o_WfdH = alloc((size_t)Jc * Kcat * 2);
  size_t o_bn   = alloc((size_t)Nc * Gc * 4);
  size_t o_bfd  = alloc((size_t)Jc * 4);
  size_t o_degf = alloc((size_t)Nc * 4);
  size_t o_degb = alloc((size_t)Nc * 4);
  size_t o_nwf  = alloc((size_t)Ec * 4);
  size_t o_nwb  = alloc((size_t)Ec * 4);

  float*  h_f32 = (float*)(w + o_h);
  float*  z1 = (float*)(w + o_z1);
  float*  z2 = (float*)(w + o_z2);
  float*  z3 = (float*)(w + o_z3);
  float*  z4 = (float*)(w + o_z4);
  half_t* Whh_h = (half_t*)(w + o_WhhH);
  half_t* W1h = (half_t*)(w + o_W1h);
  half_t* Wfd_h = (half_t*)(w + o_WfdH);
  float*  bias_node = (float*)(w + o_bn);
  float*  bias_fd = (float*)(w + o_bfd);
  float*  deg_f = (float*)(w + o_degf);
  float*  deg_b = (float*)(w + o_degb);
  float*  nw_f = (float*)(w + o_nwf);
  float*  nw_b = (float*)(w + o_nwb);

  // zero scatter accumulators every call (deterministic); h0 handled in-kernel
  hipMemsetAsync(z1, 0, (size_t)Mc * Hc * 4, stream);
  hipMemsetAsync(z2, 0, (size_t)Mc * Hc * 4, stream);
  hipMemsetAsync(z3, 0, (size_t)Mc * Hc * 4, stream);
  hipMemsetAsync(z4, 0, (size_t)Mc * Hc * 4, stream);
  hipMemsetAsync(deg_f, 0, (size_t)Nc * 4, stream);
  hipMemsetAsync(deg_b, 0, (size_t)Nc * 4, stream);

  // weight folding
  {
    int total = Gc * Hc + Gc * 32 + Jc * Kcat + Jc;
    prep_weights_k<<<(total + 255) / 256, 256, 0, stream>>>(
        Whh, enc_W, Wih, filt_W, dec_W, filt_b, dec_b, Whh_h, W1h, Wfd_h, bias_fd);
  }
  bias_node_k<<<(Nc * Gc + 255) / 256, 256, 0, stream>>>(node_emb, enc_b, Wih, bih, bhh,
                                                         bias_node);

  // GRU: all 12 timesteps in one kernel, h resident in registers/LDS per 16-row tile
  gru_all_k<<<TILES / GRU_WAVES, GRU_WAVES * 32, 0, stream>>>(
      x, W1h, bias_node, bhh, Whh_h, h_f32);

  // diffusion conv (forward 2 hops, backward 2 hops)
  deg_k<<<(Ec + 255) / 256, 256, 0, stream>>>(e_src, e_tgt, edge_w, deg_f, deg_b);
  nw_k<<<(Ec + 255) / 256, 256, 0, stream>>>(e_src, e_tgt, edge_w, deg_f, deg_b, nw_f, nw_b);
  int pgrid = (Ec * 32 + 255) / 256;
  prop_k<<<pgrid, 256, 0, stream>>>(e_src, e_tgt, nw_f, h_f32, z1);
  prop_k<<<pgrid, 256, 0, stream>>>(e_src, e_tgt, nw_f, z1, z2);
  prop_k<<<pgrid, 256, 0, stream>>>(e_tgt, e_src, nw_b, h_f32, z3);
  prop_k<<<pgrid, 256, 0, stream>>>(e_tgt, e_src, nw_b, z3, z4);

  // fused filter+decoder GEMM, writes [B,HOR,N,F] directly
  out_gemm_k<<<TILES / GRU_WAVES, GRU_WAVES * 32, 0, stream>>>(
      h_f32, z1, z2, z3, z4, Wfd_h, bias_fd, (float*)d_out);
}
